// LinearTransformerDecoder_24318104830479
// MI455X (gfx1250) — matrix operand order repaired and checked by reading, not compiled, not run on hardware
//
#include <hip/hip_runtime.h>
#include <hip/hip_bf16.h>

typedef __bf16 bf16;
typedef __attribute__((ext_vector_type(16))) __bf16 v16bf;
typedef __attribute__((ext_vector_type(8)))  __bf16 v8bf;
typedef __attribute__((ext_vector_type(8)))  float  v8f;

#define NWIN 128
#define SEQ  384
#define CH   256
#define TC   768
#define FFD  1024
#define DH   32
#define NH   8

union Frag {
    v16bf v;
    v8bf  h[2];
    unsigned short e[16];
};

__device__ __forceinline__ bf16 f2bf(float f) {
    unsigned u = __float_as_uint(f);
    u += 0x7fffu + ((u >> 16) & 1u);             // round-to-nearest-even
    unsigned short hs = (unsigned short)(u >> 16);
    bf16 r;
    __builtin_memcpy(&r, &hs, sizeof(r));
    return r;
}

// A operand 16x32 (MxK), bf16, source row-major with leading dim ld.
// Lane layout: lane = m + 16*half; elem j -> K=16*half+j.
// Two contiguous 16B loads per lane.
__device__ __forceinline__ v16bf load_fragA(const bf16* base, int ld, int lane) {
    const int m = lane & 15, half = lane >> 4;
    const bf16* p = base + (long)m * ld + 16 * half;
    Frag f;
    f.h[0] = *(const v8bf*)(p);
    f.h[1] = *(const v8bf*)(p + 8);
    return f.v;
}

// B operand 32x16 (KxN) where B[k][n] = base[n*ld + k] (row-major [N,K], e.g. weights).
// Lane = n + 16*half; elem j -> K = 16*half + j : one contiguous 32B region.
__device__ __forceinline__ v16bf load_fragB_nk(const bf16* base, int ld, int lane) {
    const int n = lane & 15, half = lane >> 4;
    const bf16* p = base + (long)n * ld + 16 * half;
    Frag f;
    f.h[0] = *(const v8bf*)(p);
    f.h[1] = *(const v8bf*)(p + 8);
    return f.v;
}

// B operand 32x16 (KxN) where B[k][n] = base[k*ld + n] (row-major [K,N], e.g. V matrix).
__device__ __forceinline__ v16bf load_fragB_kn(const bf16* base, int ld, int lane) {
    const int n = lane & 15, half = lane >> 4;
    const unsigned short* p = (const unsigned short*)(base + (long)(16 * half) * ld + n);
    Frag f;
#pragma unroll
    for (int j = 0; j < 16; ++j) f.e[j] = p[(long)j * ld];
    return f.v;
}

__device__ __forceinline__ v8f wmma_bf16(v16bf a, v16bf b, v8f c) {
    return __builtin_amdgcn_wmma_f32_16x16x32_bf16(false, a, false, b, (short)0, c,
                                                   false, false);
}

// ---------------------------------------------------------------------------
// Generic batched  Y = act(X @ W^T + bias)  with dual f32/bf16 output.
// Block: 256 threads = 8 waves as 2(M) x 4(N); wave tile 32x64; block tile 64x256.
// ---------------------------------------------------------------------------
__global__ void __launch_bounds__(256) gemm_xwT(
    const bf16* __restrict__ A, long sA, int lda,
    const bf16* __restrict__ W, int ldw,
    const float* __restrict__ bias,
    float* __restrict__ outF, bf16* __restrict__ outB,
    long sO, int ldo, int relu, int K)
{
    const int lane = threadIdx.x & 31;
    const int wv   = threadIdx.x >> 5;
    const int m0 = blockIdx.y * 64 + (wv & 1) * 32;
    const int n0 = blockIdx.x * 256 + (wv >> 1) * 64;
    const bf16* Ab = A + (long)blockIdx.z * sA;

    v8f acc[2][4];
#pragma unroll
    for (int mt = 0; mt < 2; ++mt)
#pragma unroll
        for (int nt = 0; nt < 4; ++nt)
            acc[mt][nt] = (v8f){0.f, 0.f, 0.f, 0.f, 0.f, 0.f, 0.f, 0.f};

    for (int k = 0; k < K; k += 32) {
        v16bf a0 = load_fragA(Ab + (long)m0 * lda + k, lda, lane);
        v16bf a1 = load_fragA(Ab + (long)(m0 + 16) * lda + k, lda, lane);
#pragma unroll
        for (int nt = 0; nt < 4; ++nt) {
            v16bf bw = load_fragB_nk(W + (long)(n0 + nt * 16) * ldw + k, ldw, lane);
            acc[0][nt] = wmma_bf16(a0, bw, acc[0][nt]);
            acc[1][nt] = wmma_bf16(a1, bw, acc[1][nt]);
        }
    }

    const int half = lane >> 4, lcol = lane & 15;
#pragma unroll
    for (int mt = 0; mt < 2; ++mt)
#pragma unroll
        for (int nt = 0; nt < 4; ++nt) {
            const int col = n0 + nt * 16 + lcol;
            const float bv = bias ? bias[col] : 0.f;
#pragma unroll
            for (int r = 0; r < 8; ++r) {
                const int row = m0 + mt * 16 + r + 8 * half;
                float v = acc[mt][nt][r] + bv;
                if (relu) v = fmaxf(v, 0.f);
                const long off = (long)blockIdx.z * sO + (long)row * ldo + col;
                if (outF) outF[off] = v;
                if (outB) outB[off] = f2bf(v);
            }
        }
}

// ---------------------------------------------------------------------------
// Fused attention for one (window, head, 64-query block).
// 4 waves, each owning 16 query rows; per-wave 16x384 score strip in LDS.
// qkv layout [win][s][768]: q @ col h*32, k @ 256+h*32, v @ 512+h*32.
// ---------------------------------------------------------------------------
__global__ void __launch_bounds__(128) attn_kernel(
    const bf16* __restrict__ qkv, bf16* __restrict__ outp, float scale)
{
    extern __shared__ char smem[];
    const int lane = threadIdx.x & 31;
    const int wv   = threadIdx.x >> 5;
    const int qb = blockIdx.x % (SEQ / 64);
    const int h  = (blockIdx.x / (SEQ / 64)) % NH;
    const int w  = blockIdx.x / ((SEQ / 64) * NH);
    const int m0 = qb * 64 + wv * 16;

    const bf16* base = qkv + (long)w * SEQ * TC;
    const bf16* Q  = base + h * DH;
    const bf16* Km = base + CH + h * DH;
    const bf16* Vm = base + 2 * CH + h * DH;

    float* sc = (float*)(smem + (size_t)wv * (16 * SEQ * 4));
    bf16*  Pb = (bf16*)sc;   // safe overlay: row r of Pb fits inside rows <= r of sc

    const int half = lane >> 4, lcol = lane & 15;

    // ---- scores = Q @ K^T  (K-dim = dh = 32 -> one WMMA per 16x16 tile) ----
    const v16bf aq = load_fragA(Q + (long)m0 * TC, TC, lane);
    for (int nk = 0; nk < SEQ / 16; ++nk) {
        v16bf bk = load_fragB_nk(Km + (long)(nk * 16) * TC, TC, lane);
        v8f acc = (v8f){0.f, 0.f, 0.f, 0.f, 0.f, 0.f, 0.f, 0.f};
        acc = wmma_bf16(aq, bk, acc);
#pragma unroll
        for (int r = 0; r < 8; ++r)
            sc[(r + 8 * half) * SEQ + nk * 16 + lcol] = acc[r];
    }

    // ---- wave-local softmax over each of the 16 rows ----
    for (int r = 0; r < 16; ++r) {
        float v[SEQ / 32];
        float mx = -3.0e38f;
#pragma unroll
        for (int i = 0; i < SEQ / 32; ++i) {
            v[i] = sc[r * SEQ + lane + 32 * i] * scale;
            mx = fmaxf(mx, v[i]);
        }
#pragma unroll
        for (int s = 16; s >= 1; s >>= 1) mx = fmaxf(mx, __shfl_xor(mx, s, 32));
        float sum = 0.f;
#pragma unroll
        for (int i = 0; i < SEQ / 32; ++i) { v[i] = __expf(v[i] - mx); sum += v[i]; }
#pragma unroll
        for (int s = 16; s >= 1; s >>= 1) sum += __shfl_xor(sum, s, 32);
        const float inv = 1.f / sum;
#pragma unroll
        for (int i = 0; i < SEQ / 32; ++i)
            Pb[r * SEQ + lane + 32 * i] = f2bf(v[i] * inv);
    }

    // ---- O = P @ V  (16x384 x 384x32) ----
    v8f o0 = (v8f){0.f, 0.f, 0.f, 0.f, 0.f, 0.f, 0.f, 0.f};
    v8f o1 = o0;
    for (int ks = 0; ks < SEQ / 32; ++ks) {
        v16bf ap  = load_fragA(Pb + ks * 32, SEQ, lane);
        v16bf bv0 = load_fragB_kn(Vm + (long)(ks * 32) * TC, TC, lane);
        v16bf bv1 = load_fragB_kn(Vm + (long)(ks * 32) * TC + 16, TC, lane);
        o0 = wmma_bf16(ap, bv0, o0);
        o1 = wmma_bf16(ap, bv1, o1);
    }
    bf16* ob = outp + (long)w * SEQ * CH + (long)m0 * CH + h * DH;
#pragma unroll
    for (int r = 0; r < 8; ++r) {
        const int row = r + 8 * half;
        ob[(long)row * CH + lcol]      = f2bf(o0[r]);
        ob[(long)row * CH + 16 + lcol] = f2bf(o1[r]);
    }
}

// ---------------------------------------------------------------------------
// Residual add + LayerNorm; one wave per 256-wide row, dual f32/bf16 output.
// ---------------------------------------------------------------------------
__global__ void __launch_bounds__(256) ln_kernel(
    const float* __restrict__ x, const float* __restrict__ delta,
    const float* __restrict__ g, const float* __restrict__ b,
    float* __restrict__ outF, bf16* __restrict__ outB, int rows)
{
    const int wv = threadIdx.x >> 5, lane = threadIdx.x & 31;
    const int row = blockIdx.x * 8 + wv;
    if (row >= rows) return;
    const float* xr = x + (long)row * CH;
    const float* dr = delta + (long)row * CH;
    float v[8];
    float sum = 0.f;
#pragma unroll
    for (int i = 0; i < 8; ++i) { v[i] = xr[lane * 8 + i] + dr[lane * 8 + i]; sum += v[i]; }
#pragma unroll
    for (int s = 16; s >= 1; s >>= 1) sum += __shfl_xor(sum, s, 32);
    const float mean = sum * (1.f / CH);
    float vs = 0.f;
#pragma unroll
    for (int i = 0; i < 8; ++i) { const float d = v[i] - mean; vs += d * d; }
#pragma unroll
    for (int s = 16; s >= 1; s >>= 1) vs += __shfl_xor(vs, s, 32);
    const float inv = rsqrtf(vs * (1.f / CH) + 1e-5f);
#pragma unroll
    for (int i = 0; i < 8; ++i) {
        const int c = lane * 8 + i;
        const float y = (v[i] - mean) * inv * g[c] + b[c];
        outF[(long)row * CH + c] = y;
        outB[(long)row * CH + c] = f2bf(y);
    }
}

// ---------------------------------------------------------------------------
// Gather padded windows from mem + sinusoidal PE; dual f32/bf16 output.
// ---------------------------------------------------------------------------
__global__ void __launch_bounds__(256) build_windows(
    const float* __restrict__ mem, float* __restrict__ xF, bf16* __restrict__ xB)
{
    const long idx = (long)blockIdx.x * 256 + threadIdx.x;   // NWIN*SEQ*CH
    const int ch = (int)(idx & (CH - 1));
    const int s  = (int)((idx >> 8) % SEQ);
    const int win = (int)(idx / ((long)SEQ * CH));
    const int b = win >> 5, wloc = win & 31;
    const int t = wloc * 128 + s - 128;
    float val = 0.f;
    if (t >= 0 && t < 4096) val = mem[((long)b * CH + ch) * 4096 + t];
    const int i2 = ch >> 1;
    const float div = __expf((float)(2 * i2) * (-9.210340371976184f / 256.f));
    const float ang = (float)s * div;
    val += (ch & 1) ? __cosf(ang) : __sinf(ang);
    xF[idx] = val;
    xB[idx] = f2bf(val);
}

// out[b][ch][t] = x[b*32 + t/128][128 + t%128][ch]
__global__ void __launch_bounds__(256) write_out(
    const float* __restrict__ xF, float* __restrict__ out)
{
    const long idx = (long)blockIdx.x * 256 + threadIdx.x;   // 4*256*4096
    const int t  = (int)(idx & 4095);
    const int ch = (int)((idx >> 12) & 255);
    const int b  = (int)(idx >> 20);
    const int win = b * 32 + (t >> 7);
    const int s   = 128 + (t & 127);
    out[idx] = xF[((long)win * SEQ + s) * CH + ch];
}

__global__ void __launch_bounds__(256) f32_to_bf16_k(
    const float* __restrict__ in, bf16* __restrict__ out, long n)
{
    const long i = (long)blockIdx.x * 256 + threadIdx.x;
    if (i < n) out[i] = f2bf(in[i]);
}

// ---------------------------------------------------------------------------
extern "C" void kernel_launch(void* const* d_in, const int* in_sizes, int n_in,
                              void* d_out, int out_size, void* d_ws, size_t ws_size,
                              hipStream_t stream)
{
    (void)in_sizes; (void)n_in; (void)out_size; (void)ws_size;
    const float* mem   = (const float*)d_in[0];
    const float* sa_b  = (const float*)d_in[2];
    const float* sa_ob = (const float*)d_in[4];
    const float* ca_b  = (const float*)d_in[6];
    const float* ca_ob = (const float*)d_in[8];
    const float* b1f   = (const float*)d_in[10];
    const float* b2f   = (const float*)d_in[12];
    const float* ln_g  = (const float*)d_in[13];
    const float* ln_b  = (const float*)d_in[14];
    float* out = (float*)d_out;

    // ---- workspace layout ----
    char* ws = (char*)d_ws;
    float* xF     = (float*)ws; ws += (size_t)NWIN * SEQ * CH * 4;
    float* deltaF = (float*)ws; ws += (size_t)NWIN * SEQ * CH * 4;
    bf16*  xB     = (bf16*)ws;  ws += (size_t)NWIN * SEQ * CH * 2;
    bf16*  inpB   = (bf16*)ws;  ws += (size_t)NWIN * SEQ * CH * 2;
    bf16*  qkvB   = (bf16*)ws;  ws += (size_t)NWIN * SEQ * TC * 2;
    bf16*  attB   = (bf16*)ws;  ws += (size_t)NWIN * SEQ * CH * 2;
    bf16*  hB     = qkvB;       // aliases qkv+att (both dead during FFN); exact fit
    bf16*  wBsa   = (bf16*)ws;  ws += (size_t)2 * TC * CH * 2;
    bf16*  wBca   = (bf16*)ws;  ws += (size_t)2 * TC * CH * 2;
    bf16*  wBsaow = (bf16*)ws;  ws += (size_t)2 * CH * CH * 2;
    bf16*  wBcaow = (bf16*)ws;  ws += (size_t)2 * CH * CH * 2;
    bf16*  wBw1   = (bf16*)ws;  ws += (size_t)2 * FFD * CH * 2;
    bf16*  wBw2   = (bf16*)ws;  ws += (size_t)2 * CH * FFD * 2;

    (void)hipFuncSetAttribute(reinterpret_cast<const void*>(attn_kernel),
                              hipFuncAttributeMaxDynamicSharedMemorySize, 4 * 16 * SEQ * 4);

    // ---- weight f32 -> bf16 (weights are L2-resident afterwards) ----
    auto cvt = [&](const void* src, bf16* dst, long n) {
        f32_to_bf16_k<<<(unsigned)((n + 255) / 256), 256, 0, stream>>>((const float*)src, dst, n);
    };
    cvt(d_in[1],  wBsa,   2L * TC * CH);
    cvt(d_in[5],  wBca,   2L * TC * CH);
    cvt(d_in[3],  wBsaow, 2L * CH * CH);
    cvt(d_in[7],  wBcaow, 2L * CH * CH);
    cvt(d_in[9],  wBw1,   2L * FFD * CH);
    cvt(d_in[11], wBw2,   2L * CH * FFD);

    build_windows<<<(NWIN * SEQ * CH) / 256, 256, 0, stream>>>(mem, xF, xB);

    const float scale = 0.17677669529663687f;   // 1/sqrt(32)
    const long sAct = (long)SEQ * CH;
    const int lnBlocks = (NWIN * SEQ) / 8;
    const int attnGrid = NWIN * NH * (SEQ / 64);
    const unsigned attnLds = 4 * 16 * SEQ * 4;  // 96 KB (of 320 KB per WGP)

    for (int l = 0; l < 2; ++l) {
        // keep layer input for cross-attention K/V
        hipMemcpyAsync(inpB, xB, (size_t)NWIN * SEQ * CH * 2,
                       hipMemcpyDeviceToDevice, stream);

        // -- self-attention --
        gemm_xwT<<<dim3(3, 6, NWIN), 256, 0, stream>>>(
            xB, sAct, CH, wBsa + (long)l * TC * CH, CH, sa_b + (long)l * TC,
            nullptr, qkvB, (long)SEQ * TC, TC, 0, CH);
        attn_kernel<<<attnGrid, 128, attnLds, stream>>>(qkvB, attB, scale);
        gemm_xwT<<<dim3(1, 6, NWIN), 256, 0, stream>>>(
            attB, sAct, CH, wBsaow + (long)l * CH * CH, CH, sa_ob + (long)l * CH,
            deltaF, nullptr, sAct, CH, 0, CH);
        ln_kernel<<<lnBlocks, 256, 0, stream>>>(
            xF, deltaF, ln_g + (long)(l * 3 + 0) * CH, ln_b + (long)(l * 3 + 0) * CH,
            xF, xB, NWIN * SEQ);

        // -- cross-attention (q from x, k/v from layer input) --
        gemm_xwT<<<dim3(1, 6, NWIN), 256, 0, stream>>>(
            xB, sAct, CH, wBca + (long)l * TC * CH, CH, ca_b + (long)l * TC,
            nullptr, qkvB, (long)SEQ * TC, TC, 0, CH);
        gemm_xwT<<<dim3(2, 6, NWIN), 256, 0, stream>>>(
            inpB, sAct, CH, wBca + (long)l * TC * CH + (long)CH * CH, CH,
            ca_b + (long)l * TC + CH,
            nullptr, qkvB + CH, (long)SEQ * TC, TC, 0, CH);
        attn_kernel<<<attnGrid, 128, attnLds, stream>>>(qkvB, attB, scale);
        gemm_xwT<<<dim3(1, 6, NWIN), 256, 0, stream>>>(
            attB, sAct, CH, wBcaow + (long)l * CH * CH, CH, ca_ob + (long)l * CH,
            deltaF, nullptr, sAct, CH, 0, CH);
        ln_kernel<<<lnBlocks, 256, 0, stream>>>(
            xF, deltaF, ln_g + (long)(l * 3 + 1) * CH, ln_b + (long)(l * 3 + 1) * CH,
            xF, xB, NWIN * SEQ);

        // -- FFN --
        gemm_xwT<<<dim3(4, 6, NWIN), 256, 0, stream>>>(
            xB, sAct, CH, wBw1 + (long)l * FFD * CH, CH, b1f + (long)l * FFD,
            nullptr, hB, (long)SEQ * FFD, FFD, 1, CH);
        gemm_xwT<<<dim3(1, 6, NWIN), 256, 0, stream>>>(
            hB, (long)SEQ * FFD, FFD, wBw2 + (long)l * CH * FFD, FFD, b2f + (long)l * CH,
            deltaF, nullptr, sAct, CH, 0, FFD);
        ln_kernel<<<lnBlocks, 256, 0, stream>>>(
            xF, deltaF, ln_g + (long)(l * 3 + 2) * CH, ln_b + (long)(l * 3 + 2) * CH,
            xF, xB, NWIN * SEQ);
    }

    write_out<<<(4 * 256 * 4096) / 256, 256, 0, stream>>>(xF, out);
}
